// Model_60833916780697
// MI455X (gfx1250) — matrix operand order repaired and checked by reading, not compiled, not run on hardware
//
#include <hip/hip_runtime.h>
#include <hip/hip_bf16.h>
#include <math.h>

// ---------------- problem constants ----------------
constexpr int kB    = 128;   // batch
constexpr int kL    = 192;   // encoder length
constexpr int kEnc  = 32;    // enc_in
constexpr int kCout = 32;    // c_out
constexpr int kPred = 96;    // pred_len
constexpr int kH    = 1024;  // hidden
constexpr int kG3   = 3 * kH;

// ---------------- WMMA types ----------------
typedef __attribute__((ext_vector_type(16))) __bf16 v16bf;
typedef __attribute__((ext_vector_type(8)))  __bf16 v8bf;
typedef __attribute__((ext_vector_type(8)))  float  v8f;

union Frag {
    v16bf f;
    v8bf  h[2];
};

__device__ __forceinline__ v8f wmma_bf16(const Frag& a, const Frag& b, v8f c) {
    return __builtin_amdgcn_wmma_f32_16x16x32_bf16(
        /*neg_a=*/false, a.f, /*neg_b=*/false, b.f,
        /*c_mod=*/(short)0, c, /*reuse_a=*/false, /*reuse_b=*/false);
}

__device__ __forceinline__ float sigmoidf_dev(float x) {
    return 1.0f / (1.0f + __expf(-x));
}

// ---------------- Tensor Data Mover (TDM) staging ----------------
// ROCm 7.2 clang-22: 5-arg builtin; amdgpu-toolchain clang-23 (ships the
// gfx1250 TDM header): 6-arg builtin. Detect via the header.
#if __has_include(<hip/amd_detail/amd_gfx1250_TDM.h>)
#define TDM_6ARG 1
#endif

#if defined(__has_builtin)
#if __has_builtin(__builtin_amdgcn_tensor_load_to_lds)
#define HAVE_TDM 1
#endif
#endif

typedef __attribute__((ext_vector_type(4))) unsigned int v4u;
typedef __attribute__((ext_vector_type(8))) int          v8i;
typedef __attribute__((ext_vector_type(4))) int          v4i;

#if defined(HAVE_TDM)
// 2D tile load: tile_d1 rows x tile_d0 elements (bf16), row stride in elements.
// D# packed per CDNA5 ISA 8.3/8.4: group0 = {count/type/addr}, group1 =
// {data_size, tensor dims, tile dims, dim0 stride}; groups 2/3 zero (2D).
__device__ __forceinline__ void tdm_load_tile_bf16(unsigned int lds_off,
                                                   const void* gsrc,
                                                   unsigned int tile_d0,
                                                   unsigned int tile_d1,
                                                   unsigned long long stride_e) {
    unsigned long long ga = (unsigned long long)(uintptr_t)gsrc;
    v4u g0;
    g0[0] = 1u;                                     // count=1 (valid user D#)
    g0[1] = lds_off;                                // lds_addr (bytes)
    g0[2] = (unsigned int)ga;                       // global_addr[31:0]
    g0[3] = (unsigned int)(ga >> 32) | (2u << 30);  // global_addr[56:32] | type=2
    unsigned int td0 = (unsigned int)stride_e;      // tensor_dim0 (>= tile_d0)
    unsigned int td1 = tile_d1;                     // tensor_dim1
    v8i g1;
    g1[0] = (int)(1u << 16);                                   // data_size=1 (2B)
    g1[1] = (int)((td0 & 0xFFFFu) << 16);                      // tensor_dim0 lo
    g1[2] = (int)((td0 >> 16) | ((td1 & 0xFFFFu) << 16));      // dim0 hi | dim1 lo
    g1[3] = (int)((td1 >> 16) | ((tile_d0 & 0xFFFFu) << 16));  // dim1 hi | tile_dim0
    g1[4] = (int)(tile_d1 & 0xFFFFu);                          // tile_dim1 (dim2=0)
    g1[5] = (int)(unsigned int)stride_e;                       // dim0 stride lo
    g1[6] = (int)((unsigned int)(stride_e >> 32) & 0xFFFFu);   // dim0 stride hi
    g1[7] = 0;
    v4i z4 = {0, 0, 0, 0};
#if defined(TDM_6ARG)
    v8i z8 = {0, 0, 0, 0, 0, 0, 0, 0};
    __builtin_amdgcn_tensor_load_to_lds(g0, g1, z4, z4, z8, 0);
#else
    __builtin_amdgcn_tensor_load_to_lds(g0, g1, z4, z4, 0);
#endif
}
#endif

// ---------------- fp32 -> bf16 convert ----------------
__global__ void cvt_bf16_kernel(const float* __restrict__ s, __bf16* __restrict__ d, int n) {
    for (int i = blockIdx.x * blockDim.x + threadIdx.x; i < n; i += gridDim.x * blockDim.x)
        d[i] = (__bf16)s[i];
}

// build teacher-forced decoder inputs, bf16, layout [b][t][32]
__global__ void build_dec_in_kernel(const float* __restrict__ x_enc,
                                    const float* __restrict__ x_dec,
                                    __bf16* __restrict__ d) {
    int i = blockIdx.x * blockDim.x + threadIdx.x;
    int n = kB * kPred * kCout;
    if (i >= n) return;
    int c = i & 31;
    int t = (i >> 5) % kPred;
    int b = i / (kPred * kCout);
    float v;
    if (t == 0)
        v = x_enc[(size_t)b * kL * kEnc + (size_t)(kL - 1) * kEnc + c];
    else
        v = x_dec[(size_t)b * kPred * kCout + (size_t)(t - 1) * kCout + c];
    d[i] = (__bf16)v;
}

// ---------------- fused GRU cell step ----------------
// One launch = one GRU cell for one timestep.
//   gi = x @ Wih^T          (K = kin)
//   gh = h @ Whh^T          (K = kH)
//   r = sig(gi_r+gh_r+b), z = sig(gi_z+gh_z+b), n = tanh(gi_n+b + r*(gh_n+b))
//   h' = (1-z)*n + z*h
// grid = (8 o-groups, 8 m-tiles), block = 256 (8 waves); each wave owns one
// 16x16 output tile across all three gates (6 fp32 WMMA accumulators).
// A (h and x) tiles are staged in LDS by the Tensor Data Mover.
__global__ __launch_bounds__(256)
void gru_cell_kernel(const __bf16* __restrict__ x, int x_stride, int kin,
                     const __bf16* __restrict__ hbf, const float* __restrict__ hf,
                     const __bf16* __restrict__ Wih,   // [3H][kin] row-major
                     const __bf16* __restrict__ Whh,   // [3H][H]   row-major
                     const float* __restrict__ bih, const float* __restrict__ bhh,
                     __bf16* __restrict__ hobf, float* __restrict__ hof)
{
    __shared__ __align__(16) __bf16 shH[16 * kH];
    __shared__ __align__(16) __bf16 shX[16 * kH];

    const int tid  = threadIdx.x;
    const int wid  = tid >> 5;
    const int lane = tid & 31;
    const int m0   = blockIdx.y * 16;
    const int o0   = (blockIdx.x * 8 + wid) * 16;

#if defined(HAVE_TDM)
    // ---- stage A tiles via the Tensor Data Mover (one issuing wave) ----
    if (wid == 0) {
        tdm_load_tile_bf16((unsigned int)(uintptr_t)shH,
                           hbf + (size_t)m0 * kH,
                           (unsigned int)kH, 16u, (unsigned long long)kH);
        tdm_load_tile_bf16((unsigned int)(uintptr_t)shX,
                           x + (size_t)m0 * x_stride,
                           (unsigned int)kin, 16u, (unsigned long long)x_stride);
        __builtin_amdgcn_s_wait_tensorcnt(0);
    }
    __syncthreads();
#else
    // ---- fallback: manual VMEM->LDS staging ----
    {
        const uint4* src = (const uint4*)(hbf + (size_t)m0 * kH);
        uint4* dst = (uint4*)shH;
        for (int i = tid; i < 16 * kH / 8; i += 256) dst[i] = src[i];
    }
    {
        const int rv4 = kin / 8;  // uint4 per row
        uint4* dst = (uint4*)shX;
        for (int i = tid; i < 16 * rv4; i += 256) {
            int r = i / rv4, c = i - r * rv4;
            dst[i] = ((const uint4*)(x + (size_t)(m0 + r) * x_stride))[c];
        }
    }
    __syncthreads();
#endif

    const int nloc = lane & 15;         // output column within tile / A row
    const int koff = (lane >> 4) * 8;   // unused; A and B fragments both use kbh
    const int kbh  = (lane >> 4) * 16;  // B fragment K half offset

    v8f a_ir = {}, a_iz = {}, a_in = {}, a_hr = {}, a_hz = {}, a_hn = {};

    // ---- gi = x @ Wih^T over K = kin ----
    const __bf16* wiR = Wih + (size_t)(0 * kH + o0 + nloc) * kin;
    const __bf16* wiZ = Wih + (size_t)(1 * kH + o0 + nloc) * kin;
    const __bf16* wiN = Wih + (size_t)(2 * kH + o0 + nloc) * kin;
    for (int kk = 0; kk < kin; kk += 32) {
        Frag a;
        a.h[0] = *(const v8bf*)(shX + (size_t)nloc * kin + kk + kbh);
        a.h[1] = *(const v8bf*)(shX + (size_t)nloc * kin + kk + kbh + 8);
        Frag b;
        b.h[0] = *(const v8bf*)(wiR + kk + kbh);
        b.h[1] = *(const v8bf*)(wiR + kk + kbh + 8);
        a_ir = wmma_bf16(a, b, a_ir);
        b.h[0] = *(const v8bf*)(wiZ + kk + kbh);
        b.h[1] = *(const v8bf*)(wiZ + kk + kbh + 8);
        a_iz = wmma_bf16(a, b, a_iz);
        b.h[0] = *(const v8bf*)(wiN + kk + kbh);
        b.h[1] = *(const v8bf*)(wiN + kk + kbh + 8);
        a_in = wmma_bf16(a, b, a_in);
    }

    // ---- gh = h @ Whh^T over K = kH ----
    const __bf16* whR = Whh + (size_t)(0 * kH + o0 + nloc) * kH;
    const __bf16* whZ = Whh + (size_t)(1 * kH + o0 + nloc) * kH;
    const __bf16* whN = Whh + (size_t)(2 * kH + o0 + nloc) * kH;
    for (int kk = 0; kk < kH; kk += 32) {
        __builtin_prefetch(whR + kk + 512, 0, 1);  // global_prefetch_b8
        Frag a;
        a.h[0] = *(const v8bf*)(shH + (size_t)nloc * kH + kk + kbh);
        a.h[1] = *(const v8bf*)(shH + (size_t)nloc * kH + kk + kbh + 8);
        Frag b;
        b.h[0] = *(const v8bf*)(whR + kk + kbh);
        b.h[1] = *(const v8bf*)(whR + kk + kbh + 8);
        a_hr = wmma_bf16(a, b, a_hr);
        b.h[0] = *(const v8bf*)(whZ + kk + kbh);
        b.h[1] = *(const v8bf*)(whZ + kk + kbh + 8);
        a_hz = wmma_bf16(a, b, a_hz);
        b.h[0] = *(const v8bf*)(whN + kk + kbh);
        b.h[1] = *(const v8bf*)(whN + kk + kbh + 8);
        a_hn = wmma_bf16(a, b, a_hn);
    }

    // ---- gate math + blend epilogue ----
    const int ocol = o0 + nloc;
    const float bir = bih[ocol],          bhr = bhh[ocol];
    const float biz = bih[kH + ocol],     bhz = bhh[kH + ocol];
    const float bin = bih[2 * kH + ocol], bhn = bhh[2 * kH + ocol];
    const int rbase = m0 + ((lane >> 4) ? 8 : 0);
#pragma unroll
    for (int i = 0; i < 8; ++i) {
        const int m = rbase + i;
        const size_t idx = (size_t)m * kH + ocol;
        float r  = sigmoidf_dev(a_ir[i] + a_hr[i] + bir + bhr);
        float z  = sigmoidf_dev(a_iz[i] + a_hz[i] + biz + bhz);
        float nn = tanhf(a_in[i] + bin + r * (a_hn[i] + bhn));
        float hp = hf[idx];
        float hv = (1.0f - z) * nn + z * hp;
        hof[idx]  = hv;
        hobf[idx] = (__bf16)hv;
    }
}

// ---------------- output projection ----------------
// out[b][t][o] = ys[t][b][:] . out_W[o][:] + out_b[o]
// ys: [kPred*kB][kH] bf16 row-major. grid = 768 M-tiles, block = 64 (2 waves = 2 N-tiles).
__global__ __launch_bounds__(64)
void out_proj_kernel(const __bf16* __restrict__ ys, const __bf16* __restrict__ Wo,
                     const float* __restrict__ bo, float* __restrict__ out)
{
    const int wid  = threadIdx.x >> 5;
    const int lane = threadIdx.x & 31;
    const int m0   = blockIdx.x * 16;
    const int o0   = wid * 16;
    const int nloc = lane & 15;
    const int koff = (lane >> 4) * 8;
    const int kbh  = (lane >> 4) * 16;

    const __bf16* arow = ys + (size_t)(m0 + nloc) * kH;
    const __bf16* brow = Wo + (size_t)(o0 + nloc) * kH;
    v8f acc = {};
    for (int kk = 0; kk < kH; kk += 32) {
        Frag a;
        a.h[0] = *(const v8bf*)(arow + kk + kbh);
        a.h[1] = *(const v8bf*)(arow + kk + kbh + 8);
        Frag b;
        b.h[0] = *(const v8bf*)(brow + kk + kbh);
        b.h[1] = *(const v8bf*)(brow + kk + kbh + 8);
        acc = wmma_bf16(a, b, acc);
    }
    const float bb = bo[o0 + nloc];
    const int rbase = m0 + ((lane >> 4) ? 8 : 0);
#pragma unroll
    for (int i = 0; i < 8; ++i) {
        const int m = rbase + i;        // m = t*kB + b
        const int t = m >> 7;
        const int b = m & 127;
        out[(size_t)b * kPred * kCout + (size_t)t * kCout + o0 + nloc] = acc[i] + bb;
    }
}

// ---------------- host orchestration ----------------
extern "C" void kernel_launch(void* const* d_in, const int* in_sizes, int n_in,
                              void* d_out, int out_size, void* d_ws, size_t ws_size,
                              hipStream_t stream) {
    (void)in_sizes; (void)n_in; (void)out_size; (void)ws_size;

    const float* x_enc = (const float*)d_in[0];
    const float* x_dec = (const float*)d_in[2];
    const float* eWih0 = (const float*)d_in[4];
    const float* eWhh0 = (const float*)d_in[5];
    const float* ebih0 = (const float*)d_in[6];
    const float* ebhh0 = (const float*)d_in[7];
    const float* eWih1 = (const float*)d_in[8];
    const float* eWhh1 = (const float*)d_in[9];
    const float* ebih1 = (const float*)d_in[10];
    const float* ebhh1 = (const float*)d_in[11];
    const float* dWih0 = (const float*)d_in[12];
    const float* dWhh0 = (const float*)d_in[13];
    const float* dbih0 = (const float*)d_in[14];
    const float* dbhh0 = (const float*)d_in[15];
    const float* dWih1 = (const float*)d_in[16];
    const float* dWhh1 = (const float*)d_in[17];
    const float* dbih1 = (const float*)d_in[18];
    const float* dbhh1 = (const float*)d_in[19];
    const float* outW  = (const float*)d_in[20];
    const float* outB  = (const float*)d_in[21];
    float* out = (float*)d_out;

    // ---- workspace bump allocator (256B aligned) ----
    char* p = (char*)d_ws;
    auto alloc = [&](size_t bytes) -> void* {
        void* r = (void*)p;
        p += (bytes + 255) & ~(size_t)255;
        return r;
    };
    const size_t nBig = (size_t)kG3 * kH;   // 3072*1024
    const size_t nSml = (size_t)kG3 * kEnc; // 3072*32
    const size_t nSt  = (size_t)kB * kH;    // 128*1024

    __bf16* wEih0 = (__bf16*)alloc(nSml * 2);
    __bf16* wEhh0 = (__bf16*)alloc(nBig * 2);
    __bf16* wEih1 = (__bf16*)alloc(nBig * 2);
    __bf16* wEhh1 = (__bf16*)alloc(nBig * 2);
    __bf16* wDih0 = (__bf16*)alloc(nSml * 2);
    __bf16* wDhh0 = (__bf16*)alloc(nBig * 2);
    __bf16* wDih1 = (__bf16*)alloc(nBig * 2);
    __bf16* wDhh1 = (__bf16*)alloc(nBig * 2);
    __bf16* wOut  = (__bf16*)alloc((size_t)kCout * kH * 2);
    __bf16* xeBf  = (__bf16*)alloc((size_t)kB * kL * kEnc * 2);
    __bf16* diBf  = (__bf16*)alloc((size_t)kB * kPred * kCout * 2);
    __bf16* h0bf[2] = { (__bf16*)alloc(nSt * 2), (__bf16*)alloc(nSt * 2) };
    __bf16* h1bf[2] = { (__bf16*)alloc(nSt * 2), (__bf16*)alloc(nSt * 2) };
    __bf16* ys    = (__bf16*)alloc((size_t)kPred * nSt * 2);
    float*  h0f[2] = { (float*)alloc(nSt * 4), (float*)alloc(nSt * 4) };
    float*  h1f[2] = { (float*)alloc(nSt * 4), (float*)alloc(nSt * 4) };

    // ---- weight / input conversion to bf16 ----
    auto cvt = [&](const float* s, __bf16* d, size_t n) {
        int blocks = (int)((n + 255) / 256);
        cvt_bf16_kernel<<<blocks, 256, 0, stream>>>(s, d, (int)n);
    };
    cvt(eWih0, wEih0, nSml); cvt(eWhh0, wEhh0, nBig);
    cvt(eWih1, wEih1, nBig); cvt(eWhh1, wEhh1, nBig);
    cvt(dWih0, wDih0, nSml); cvt(dWhh0, wDhh0, nBig);
    cvt(dWih1, wDih1, nBig); cvt(dWhh1, wDhh1, nBig);
    cvt(outW,  wOut,  (size_t)kCout * kH);
    cvt(x_enc, xeBf,  (size_t)kB * kL * kEnc);
    {
        int n = kB * kPred * kCout;
        build_dec_in_kernel<<<(n + 255) / 256, 256, 0, stream>>>(x_enc, x_dec, diBf);
    }

    // ---- zero initial hidden state (index 0 of each ping-pong pair) ----
    hipMemsetAsync(h0f[0], 0, nSt * 4, stream);
    hipMemsetAsync(h1f[0], 0, nSt * 4, stream);
    hipMemsetAsync(h0bf[0], 0, nSt * 2, stream);
    hipMemsetAsync(h1bf[0], 0, nSt * 2, stream);

    const dim3 cgrid(8, 8, 1);   // 8 o-groups x 8 m-tiles

    // ---- encoder: 192 steps x 2 layers ----
    for (int t = 0; t < kL; ++t) {
        const int pi = t & 1, po = pi ^ 1;
        gru_cell_kernel<<<cgrid, 256, 0, stream>>>(
            xeBf + (size_t)t * kEnc, kL * kEnc, kEnc,
            h0bf[pi], h0f[pi], wEih0, wEhh0, ebih0, ebhh0, h0bf[po], h0f[po]);
        gru_cell_kernel<<<cgrid, 256, 0, stream>>>(
            h0bf[po], kH, kH,
            h1bf[pi], h1f[pi], wEih1, wEhh1, ebih1, ebhh1, h1bf[po], h1f[po]);
    }

    // ---- decoder: 96 steps x 2 layers (kL even -> states land at index 0) ----
    for (int t = 0; t < kPred; ++t) {
        const int pi = t & 1, po = pi ^ 1;
        gru_cell_kernel<<<cgrid, 256, 0, stream>>>(
            diBf + (size_t)t * kCout, kPred * kCout, kCout,
            h0bf[pi], h0f[pi], wDih0, wDhh0, dbih0, dbhh0, h0bf[po], h0f[po]);
        const __bf16* h1in = (t == 0) ? h1bf[0] : (ys + (size_t)(t - 1) * nSt);
        gru_cell_kernel<<<cgrid, 256, 0, stream>>>(
            h0bf[po], kH, kH,
            h1in, h1f[pi], wDih1, wDhh1, dbih1, dbhh1,
            ys + (size_t)t * nSt, h1f[po]);
    }

    // ---- output projection: [12288 x 1024] @ [1024 x 32] + bias ----
    out_proj_kernel<<<(kPred * kB) / 16, 64, 0, stream>>>(ys, wOut, outB, out);
}
